// PolicyNet_user_80487687127524
// MI455X (gfx1250) — compile-verified, matrix operand order checked by reading, not run on hardware
//
#include <hip/hip_runtime.h>
#include <hip/hip_bf16.h>
#include <math.h>

typedef __attribute__((ext_vector_type(16))) _Float16 v16h;
typedef __attribute__((ext_vector_type(8)))  _Float16 v8h;
typedef __attribute__((ext_vector_type(8)))  float    v8f;

#define B_   32
#define N_   256
#define L_   50
#define DU_  32
#define DI_  32
#define DC_  16
#define D_   48     // DI + DC
#define FD_  192    // 4*D
#define H1_  64
#define H2_  16
#define LP_  64     // L padded to WMMA tile multiple
#define TOT_ 128
#define F1_  200
#define F2_  80

// Fragment helpers ----------------------------------------------------------
// 16-bit A/B fragment element j of lane ln covers k_local =
//   (p<4 ? kh+2p : 16+kh+2(p-4)) + o   with p=j>>1, o=j&1, kh=(ln>>4)*8.
// For a row-contiguous f16 buffer this is just rows[kb+kh+j] (j<8) and
// rows[kb+16+kh+j-8] (j>=8)  ->  two 16-byte LDS loads.
__device__ __forceinline__ v16h load_b_frag_row(const _Float16* row, int kb, int kh)
{
    v8h lo = *(const v8h*)(row + kb + kh);
    v8h hi = *(const v8h*)(row + kb + 16 + kh);
    return __builtin_shufflevector(lo, hi, 0, 1, 2, 3, 4, 5, 6, 7,
                                           8, 9, 10, 11, 12, 13, 14, 15);
}

__device__ __forceinline__ v16h load_frag_lds(const _Float16* fragbase, int lane)
{
    const _Float16* p = fragbase + lane * 16;
    v8h lo = *(const v8h*)(p);
    v8h hi = *(const v8h*)(p + 8);
    return __builtin_shufflevector(lo, hi, 0, 1, 2, 3, 4, 5, 6, 7,
                                           8, 9, 10, 11, 12, 13, 14, 15);
}

// One workgroup = one (b,n) pair. D = W^T x ai^T so weights live in registers
// as A fragments and activations stream as contiguous-row B fragments.
__global__ __launch_bounds__(128)
void din_attention_kernel(const float* __restrict__ emb_user,
                          const float* __restrict__ emb_item,
                          const float* __restrict__ emb_cate,
                          const float* __restrict__ att_w1, const float* __restrict__ att_b1,
                          const float* __restrict__ att_w2, const float* __restrict__ att_b2,
                          const float* __restrict__ att_w3, const float* __restrict__ att_b3,
                          const float* __restrict__ fc_w1,  const float* __restrict__ fc_b1,
                          const float* __restrict__ fc_w2,  const float* __restrict__ fc_b2,
                          const float* __restrict__ fc_w3,  const float* __restrict__ fc_b3,
                          const int* __restrict__ user,
                          const int* __restrict__ items,
                          const int* __restrict__ items_cate,
                          const int* __restrict__ interest_items,
                          const int* __restrict__ interest_items_cate,
                          float* __restrict__ logits)
{
    // W1^T as 4 mtiles x 6 ktiles of pre-swizzled A fragments (512 halves each)
    __shared__ __align__(16) _Float16 w1f[24 * 512];   // 24576 B
    __shared__ __align__(16) _Float16 w2f[2 * 512];    //  2048 B
    __shared__ __align__(16) _Float16 ai16[LP_ * FD_]; // [64][192] f16  24576 B
    __shared__ __align__(16) _Float16 h1s[LP_ * H1_];  // [row][feat]     8192 B
    __shared__ float qs[D_];
    __shared__ float scores[LP_];
    __shared__ float feat[TOT_];
    __shared__ float x1[F1_];
    __shared__ float x2[F2_];
    __shared__ float red[128];

    const int bn   = blockIdx.x;
    const int tid  = threadIdx.x;
    const int lane = tid & 31;
    const int wave = tid >> 5;

    // ---- stage W1^T / W2^T into LDS in per-lane fragment layout ----
    for (int e = tid; e < 24 * 512; e += 128) {
        int f = e >> 9, r = e & 511;
        int ln = r >> 4, j = r & 15;
        int mt = f / 6, kt = f - mt * 6;
        int m  = ln & 15;
        int kh = (ln >> 4) << 3;
        int p  = j >> 1, o = j & 1;
        int kl = ((p < 4) ? (kh + 2 * p) : (16 + kh + 2 * (p - 4))) + o;
        w1f[e] = (_Float16)att_w1[(kt * 32 + kl) * H1_ + mt * 16 + m];
    }
    for (int e = tid; e < 2 * 512; e += 128) {
        int kt = e >> 9, r = e & 511;
        int ln = r >> 4, j = r & 15;
        int m  = ln & 15;
        int kh = (ln >> 4) << 3;
        int p  = j >> 1, o = j & 1;
        int kl = ((p < 4) ? (kh + 2 * p) : (16 + kh + 2 * (p - 4))) + o;
        w2f[e] = (_Float16)att_w2[(kt * 32 + kl) * H2_ + m];
    }

    // ---- query vector + user embedding ----
    if (tid < D_) {
        float v = (tid < DI_) ? emb_item[(long)items[bn] * DI_ + tid]
                              : emb_cate[(long)items_cate[bn] * DC_ + (tid - DI_)];
        qs[tid]   = v;
        feat[tid] = v;
    }
    if (tid < DU_)
        feat[2 * D_ + tid] = emb_user[(long)user[bn] * DU_ + tid];

    // ---- gather he into quad-1 of ai16 (cols 48..95), pad rows zero ----
    const long hbase = (long)bn * L_;
    for (int e = tid; e < LP_ * D_; e += 128) {
        int l = e / D_, d = e - l * D_;
        float v = 0.f;
        if (l < L_) {
            v = (d < DI_) ? emb_item[(long)interest_items[hbase + l] * DI_ + d]
                          : emb_cate[(long)interest_items_cate[hbase + l] * DC_ + (d - DI_)];
        }
        ai16[l * FD_ + D_ + d] = (_Float16)v;
    }
    __syncthreads();

    // ---- fill quads 0 (q), 2 (q-k), 3 (q*k) ----
    for (int e = tid; e < LP_ * D_; e += 128) {
        int l = e / D_, d = e - l * D_;
        float hv = (float)ai16[l * FD_ + D_ + d];
        float qv = qs[d];
        ai16[l * FD_ + d]           = (_Float16)qv;
        ai16[l * FD_ + 2 * D_ + d]  = (_Float16)(qv - hv);
        ai16[l * FD_ + 3 * D_ + d]  = (_Float16)(qv * hv);
    }
    __syncthreads();

    const int nidx = lane & 15;           // N index = row-in-tile
    const int kh   = (lane >> 4) << 3;
    const int hi   = lane >> 4;           // 0/1 -> feature sub-block

    // ---- hoist this wave's W1^T A-fragments into registers ----
    v16h a1[6];
#pragma unroll
    for (int kt = 0; kt < 6; ++kt)
        a1[kt] = load_frag_lds(&w1f[(wave * 6 + kt) * 512], lane);
    v16h a2[2];
#pragma unroll
    for (int kt = 0; kt < 2; ++kt)
        a2[kt] = load_frag_lds(&w2f[kt * 512], lane);

    // ========== Layer 1: D[feat][row] = W1^T x ai^T, ReLU, -> h1s[row][feat]
#pragma unroll
    for (int nt = 0; nt < 4; ++nt) {
        const int l = nt * 16 + nidx;
        const _Float16* airow = &ai16[l * FD_];
        v8f acc = {};
#pragma unroll
        for (int kt = 0; kt < 6; ++kt) {
            v16h b = load_b_frag_row(airow, kt * 32, kh);
            acc = __builtin_amdgcn_wmma_f32_16x16x32_f16(
                      false, a1[kt], false, b, (short)0, acc, false, false);
        }
        const float* bp = att_b1 + wave * 16 + hi * 8;   // 8 consecutive features
        v8h hpack;
#pragma unroll
        for (int r = 0; r < 8; ++r) {
            float v = acc[r] + bp[r];
            hpack[r] = (_Float16)(v > 0.f ? v : 0.f);
        }
        *(v8h*)&h1s[l * H1_ + wave * 16 + hi * 8] = hpack;   // one b128 store
    }
    __syncthreads();

    // ========== Layer 2: D[feat2][row] = W2^T x h1^T, fused w3 dot ==========
    {
        const int l = wave * 16 + nidx;                 // this wave owns rows nt=wave
        const _Float16* h1row = &h1s[l * H1_];
        v8f acc = {};
#pragma unroll
        for (int kt = 0; kt < 2; ++kt) {
            v16h b = load_b_frag_row(h1row, kt * 32, kh);
            acc = __builtin_amdgcn_wmma_f32_16x16x32_f16(
                      false, a2[kt], false, b, (short)0, acc, false, false);
        }
        // lane holds features hi*8 .. hi*8+7 for row l
        float partial = 0.f;
#pragma unroll
        for (int r = 0; r < 8; ++r) {
            int f = hi * 8 + r;
            partial += (acc[r] + att_b2[f]) * att_w3[f];
        }
        partial += __shfl_xor(partial, 16, 32);          // combine feature halves
        float s = partial + att_b3[0];
        bool valid = (l < L_) && (interest_items[hbase + l] != 0);
        if (lane < 16) scores[l] = valid ? s : 0.f;
    }
    __syncthreads();

    // ---- pooled[d] = sum_l score[l] * he[l][d] (he = quad-1 of ai16) ----
    if (tid < D_) {
        float p = 0.f;
        for (int l = 0; l < LP_; ++l)
            p += scores[l] * (float)ai16[l * FD_ + D_ + tid];
        feat[D_ + tid] = p;
    }
    __syncthreads();

    // ---- FC head: 128 -> 200 -> 80 -> 1 ----
    for (int j = tid; j < F1_; j += 128) {
        float s = fc_b1[j];
        for (int i = 0; i < TOT_; ++i) s += feat[i] * fc_w1[i * F1_ + j];
        x1[j] = s > 0.f ? s : 0.f;
    }
    __syncthreads();
    if (tid < F2_) {
        float s = fc_b2[tid];
        for (int i = 0; i < F1_; ++i) s += x1[i] * fc_w2[i * F2_ + tid];
        x2[tid] = s > 0.f ? s : 0.f;
    }
    __syncthreads();
    red[tid] = (tid < F2_) ? x2[tid] * fc_w3[tid] : 0.f;
    __syncthreads();
    for (int s = 64; s > 0; s >>= 1) {
        if (tid < s) red[tid] += red[tid + s];
        __syncthreads();
    }
    if (tid == 0) {
        float z   = red[0] + fc_b3[0];
        float sig = 1.f / (1.f + __expf(-z));
        logits[bn] = (items[bn] == 0) ? -__builtin_inff() : sig;
    }
}

// Per-batch-row softmax over N=256.
__global__ __launch_bounds__(256)
void softmax_kernel(const float* __restrict__ logits, float* __restrict__ out)
{
    __shared__ float red[256];
    const int b = blockIdx.x, t = threadIdx.x;
    float v = logits[b * N_ + t];
    red[t] = v;
    __syncthreads();
    for (int s = 128; s > 0; s >>= 1) {
        if (t < s) red[t] = fmaxf(red[t], red[t + s]);
        __syncthreads();
    }
    float mx = red[0];
    __syncthreads();
    float e = __expf(v - mx);
    red[t] = e;
    __syncthreads();
    for (int s = 128; s > 0; s >>= 1) {
        if (t < s) red[t] += red[t + s];
        __syncthreads();
    }
    out[b * N_ + t] = e / red[0];
}

extern "C" void kernel_launch(void* const* d_in, const int* in_sizes, int n_in,
                              void* d_out, int out_size, void* d_ws, size_t ws_size,
                              hipStream_t stream)
{
    const float* emb_user = (const float*)d_in[0];
    const float* emb_item = (const float*)d_in[1];
    const float* emb_cate = (const float*)d_in[2];
    const float* att_w1   = (const float*)d_in[3];
    const float* att_b1   = (const float*)d_in[4];
    const float* att_w2   = (const float*)d_in[5];
    const float* att_b2   = (const float*)d_in[6];
    const float* att_w3   = (const float*)d_in[7];
    const float* att_b3   = (const float*)d_in[8];
    const float* fc_w1    = (const float*)d_in[9];
    const float* fc_b1    = (const float*)d_in[10];
    const float* fc_w2    = (const float*)d_in[11];
    const float* fc_b2    = (const float*)d_in[12];
    const float* fc_w3    = (const float*)d_in[13];
    const float* fc_b3    = (const float*)d_in[14];
    const int* user                = (const int*)d_in[15];
    const int* items               = (const int*)d_in[16];
    const int* items_cate          = (const int*)d_in[17];
    const int* interest_items      = (const int*)d_in[18];
    const int* interest_items_cate = (const int*)d_in[19];

    float* logits = (float*)d_ws;   // B*N f32 scratch

    din_attention_kernel<<<B_ * N_, 128, 0, stream>>>(
        emb_user, emb_item, emb_cate,
        att_w1, att_b1, att_w2, att_b2, att_w3, att_b3,
        fc_w1, fc_b1, fc_w2, fc_b2, fc_w3, fc_b3,
        user, items, items_cate, interest_items, interest_items_cate,
        logits);

    softmax_kernel<<<B_, 256, 0, stream>>>(logits, (float*)d_out);
}
